// TriangleMultiplicativeUpdate_64527588655420
// MI455X (gfx1250) — matrix operand order repaired and checked by reading, not compiled, not run on hardware
//
#include <hip/hip_runtime.h>
#include <hip/hip_bf16.h>

// ---------------------------------------------------------------------------
// TriangleMultiplicativeUpdate (outgoing), B=1, N=512, C_Z=C_H=128, f32 I/O.
// Strategy: bf16 WMMA (v_wmma_f32_16x16x32_bf16) for all matmuls; a/b stored
// transposed [c][row] in bf16 so the triangle einsum is 128 channel-batched
// 512x512x512 GEMMs with contiguous-K fragment loads (L2-resident: 134 MB).
// Workspace layout (bytes):
//   [0        ) wT   : 6 x 128x128 bf16, [out][in]          (196,608)
//   [262144   ) aT   : bf16 [128][512*512]                  (67,108,864)
//   [67371008 ) bT   : bf16 [128][512*512]                  (67,108,864)
//   [134479872) g    : f32  [512*512][128]                  (134,217,728)
//   total required ws >= 268,697,600 bytes
// ---------------------------------------------------------------------------

typedef __bf16 v16bf __attribute__((ext_vector_type(16)));
typedef __bf16 v8bf  __attribute__((ext_vector_type(8)));
typedef float  v8f   __attribute__((ext_vector_type(8)));

__device__ __forceinline__ v16bf mk16(v8bf lo, v8bf hi) {
    union { v16bf v; v8bf h[2]; } u;
    u.h[0] = lo; u.h[1] = hi;
    return u.v;
}

// A-matrix 16x32 bf16 fragment: lane L holds row M=L%16;
// elems 0..15 = K kb+2h..kb+2h+15 (contiguous K, matches B),
// with h = 8*(L>=16).  rowbase must point at the lane's row, 16B aligned.
__device__ __forceinline__ v16bf load_afrag(const __bf16* rowbase, int kb, int h) {
    v8bf a0 = *reinterpret_cast<const v8bf*>(rowbase + kb + 2 * h);
    v8bf a1 = *reinterpret_cast<const v8bf*>(rowbase + kb + 2 * h + 8);
    return mk16(a0, a1);
}

// B-matrix 32x16 bf16 fragment: lane L holds column N=L%16;
// elems 0..15 = K kb+hb .. kb+hb+15 with hb = 16*(L>=16) (contiguous K).
__device__ __forceinline__ v16bf load_bfrag(const __bf16* colbase, int kb, int hb) {
    v8bf b0 = *reinterpret_cast<const v8bf*>(colbase + kb + hb);
    v8bf b1 = *reinterpret_cast<const v8bf*>(colbase + kb + hb + 8);
    return mk16(b0, b1);
}

__device__ __forceinline__ v8f wmma_bf16(v16bf A, v16bf B, v8f C) {
    return __builtin_amdgcn_wmma_f32_16x16x32_bf16(
        /*neg_a=*/false, A, /*neg_b=*/false, B,
        /*c_mod=*/(short)0, C, /*reuse_a=*/false, /*reuse_b=*/false);
}

// Fast sigmoid: v_exp_f32 + v_rcp_f32 (avoids the IEEE v_div_scale/div_fmas
// sequence the compiler emits for 1.0f/x).
__device__ __forceinline__ float sigmoidf_(float x) {
    return __builtin_amdgcn_rcpf(1.0f + __expf(-x));
}

// ---------------------------------------------------------------------------
// Kernel 0: transpose+convert six [in=128][out=128] f32 weights -> bf16 [out][in]
// ---------------------------------------------------------------------------
__global__ void k_prep(const float* __restrict__ s0, const float* __restrict__ s1,
                       const float* __restrict__ s2, const float* __restrict__ s3,
                       const float* __restrict__ s4, const float* __restrict__ s5,
                       __bf16* __restrict__ dst) {
    const float* srcs[6] = { s0, s1, s2, s3, s4, s5 };
    const float* S = srcs[blockIdx.x];
    __bf16* D = dst + (size_t)blockIdx.x * 128 * 128;
    for (int e = threadIdx.x; e < 128 * 128; e += blockDim.x) {
        int o = e >> 7, i = e & 127;
        D[e] = (__bf16)S[i * 128 + o];   // D[o][i] = W[i][o]
    }
}

// ---------------------------------------------------------------------------
// Kernel 1: fused LayerNorm(in) + {a,b,g} projections + sigmoid gating + mask.
// Grid: 16384 blocks x 256 threads; each block = 16 rows of z ([262144][128]).
// Stores a,b transposed bf16 [c][row]; g as f32 [row][c].
// ---------------------------------------------------------------------------
__global__ __launch_bounds__(256) void k_ln_proj(
    const float* __restrict__ z,     const float* __restrict__ mask,
    const float* __restrict__ b_a_p, const float* __restrict__ b_a_g,
    const float* __restrict__ b_b_p, const float* __restrict__ b_b_g,
    const float* __restrict__ b_gv,
    const float* __restrict__ ln_s,  const float* __restrict__ ln_b,
    const __bf16* __restrict__ wT,   // 6 matrices, bf16 [out][in]
    __bf16* __restrict__ aT, __bf16* __restrict__ bT, float* __restrict__ gws)
{
    constexpr int C = 128;
    constexpr size_t NN = 512 * 512;
    constexpr int ZNSTR = 136;                 // padded bf16 row stride (272B, 16B-mult)

    __shared__ float red_s[16][16], red_q[16][16];
    __shared__ float mu_s[16], rs_s[16];
    __shared__ __align__(16) __bf16 zn[16 * ZNSTR];
    __shared__ __align__(16) __bf16 stage[8][16][16];

    const int t    = threadIdx.x;
    const int row  = t >> 4, sub = t & 15;
    const int row0 = blockIdx.x * 16;

    // ---- Phase A: LayerNorm over C for 16 rows (16 threads per row) ----
    float zv[8];
    {
        const size_t gbase = (size_t)(row0 + row) * C + sub * 8;
        float s = 0.f, q = 0.f;
#pragma unroll
        for (int e = 0; e < 8; ++e) { float v = z[gbase + e]; zv[e] = v; s += v; q += v * v; }
        red_s[row][sub] = s; red_q[row][sub] = q;
    }
    __syncthreads();
    if (sub == 0) {
        float ss = 0.f, qq = 0.f;
        for (int e = 0; e < 16; ++e) { ss += red_s[row][e]; qq += red_q[row][e]; }
        float mu  = ss * (1.0f / 128.0f);
        float var = qq * (1.0f / 128.0f) - mu * mu;
        mu_s[row] = mu;
        rs_s[row] = rsqrtf(var + 1e-5f);
    }
    __syncthreads();
    {
        float mu = mu_s[row], rs = rs_s[row];
#pragma unroll
        for (int e = 0; e < 8; ++e) {
            int c = sub * 8 + e;
            zn[row * ZNSTR + c] = (__bf16)((zv[e] - mu) * rs * ln_s[c] + ln_b[c]);
        }
    }
    __syncthreads();

    // ---- Phase B: WMMA projections. wave w owns output cols [16w,16w+16) ----
    const int w    = t >> 5, lane = t & 31;
    const int lM   = lane & 15, hi = lane >> 4;
    const int hA   = hi * 8, hB = hi * 16;
    const int nt   = w;
    const int c    = nt * 16 + lM;
    const __bf16* znrow = zn + lM * ZNSTR;

    v16bf Af[4];                                // K=128 -> 4 chunks, reused by all tasks
#pragma unroll
    for (int ks = 0; ks < 4; ++ks) Af[ks] = load_afrag(znrow, ks * 32, hA);

    // tasks 0/1: a and b (gated projections), transposed bf16 store
#pragma unroll
    for (int kind = 0; kind < 2; ++kind) {
        const __bf16* WpCol = wT + (size_t)(kind * 2 + 0) * 16384 + (size_t)c * 128;
        const __bf16* WgCol = wT + (size_t)(kind * 2 + 1) * 16384 + (size_t)c * 128;
        v8f accP = {}, accG = {};
#pragma unroll
        for (int ks = 0; ks < 4; ++ks) {
            accP = wmma_bf16(Af[ks], load_bfrag(WpCol, ks * 32, hB), accP);
            accG = wmma_bf16(Af[ks], load_bfrag(WgCol, ks * 32, hB), accG);
        }
        const float bp = (kind == 0 ? b_a_p : b_b_p)[c];
        const float bg = (kind == 0 ? b_a_g : b_b_g)[c];
#pragma unroll
        for (int r = 0; r < 8; ++r) {
            int M = r + hi * 8;
            float mval = mask[row0 + M];
            float val  = sigmoidf_(accG[r] + bg) * mval * (accP[r] + bp);
            stage[w][lM][M] = (__bf16)val;      // transpose to [c_local][row_local]
        }
        __builtin_amdgcn_wave_barrier();        // DS ops are in-order per wave
        if (lane < 16) {
            __bf16* dst = (kind == 0 ? aT : bT) + (size_t)(nt * 16 + lane) * NN + row0;
            v8bf s0 = *reinterpret_cast<const v8bf*>(&stage[w][lane][0]);
            v8bf s1 = *reinterpret_cast<const v8bf*>(&stage[w][lane][8]);
            *reinterpret_cast<v8bf*>(dst)     = s0;
            *reinterpret_cast<v8bf*>(dst + 8) = s1;
        }
        __builtin_amdgcn_wave_barrier();
    }

    // task 2: output gate g = sigmoid(zn @ w_g + b_g), f32 [row][c]
    {
        const __bf16* WCol = wT + (size_t)4 * 16384 + (size_t)c * 128;
        v8f acc = {};
#pragma unroll
        for (int ks = 0; ks < 4; ++ks)
            acc = wmma_bf16(Af[ks], load_bfrag(WCol, ks * 32, hB), acc);
        const float bgc = b_gv[c];
#pragma unroll
        for (int r = 0; r < 8; ++r) {
            int M = r + hi * 8;
            gws[(size_t)(row0 + M) * C + c] = sigmoidf_(acc[r] + bgc);
        }
    }
}

// ---------------------------------------------------------------------------
// Kernel 2: fused triangle einsum + LayerNorm(out) + (x@w_z + b_z) * g.
// Grid: 32x32 blocks (16x16 (i,j) tile each) x 256 threads.
// Dynamic LDS: x[128][256] f32 (128KB) + xh[256][136] bf16 (68KB) + ln (1KB).
// ---------------------------------------------------------------------------
__global__ __launch_bounds__(256) void k_tri(
    const __bf16* __restrict__ aT, const __bf16* __restrict__ bT,
    const float*  __restrict__ gws, const __bf16* __restrict__ wzT,
    const float*  __restrict__ b_z,
    const float*  __restrict__ ln_s, const float* __restrict__ ln_b,
    float* __restrict__ out)
{
    constexpr int N = 512, C = 128, XHS = 136;
    extern __shared__ char smem[];
    float*  xls  = reinterpret_cast<float*>(smem);                       // [128][256]
    __bf16* xh   = reinterpret_cast<__bf16*>(smem + 128 * 256 * 4);      // [256][136]
    float*  lnsS = reinterpret_cast<float*>(smem + 128 * 256 * 4 + 256 * XHS * 2);
    float*  lnbS = lnsS + 128;

    const int t  = threadIdx.x;
    const int w  = t >> 5, lane = t & 31, lM = lane & 15, hi = lane >> 4;
    const int hA = hi * 8, hB = hi * 16;
    const int i0 = blockIdx.y * 16, j0 = blockIdx.x * 16;

    if (t < 128) { lnsS[t] = ln_s[t]; lnbS[t] = ln_b[t]; }

    // ---- Phase 1: x[i,j,c] = sum_k a[i,k,c]*b[j,k,c]; wave w owns 16 channels ----
#pragma unroll 1
    for (int cc = 0; cc < 16; ++cc) {
        const int c = w * 16 + cc;
        const __bf16* Ab = aT + ((size_t)c * N + (i0 + lM)) * N;  // row i, contiguous k
        const __bf16* Bb = bT + ((size_t)c * N + (j0 + lM)) * N;  // row j, contiguous k
        v8f acc = {};
#pragma unroll
        for (int ks = 0; ks < 16; ++ks) {
            v16bf A = load_afrag(Ab, ks * 32, hA);
            v16bf B = load_bfrag(Bb, ks * 32, hB);
            acc = wmma_bf16(A, B, acc);
        }
#pragma unroll
        for (int r = 0; r < 8; ++r) {
            int pos = (r + hi * 8) * 16 + lM;   // pos = (i-i0)*16 + (j-j0)
            xls[c * 256 + pos] = acc[r];
        }
    }
    __syncthreads();

    // ---- Phase 2: per-position LayerNorm over 128 channels -> bf16 A-operand ----
    {
        float s = 0.f, q = 0.f;
        for (int c2 = 0; c2 < 128; ++c2) { float v = xls[c2 * 256 + t]; s += v; q += v * v; }
        float mu = s * (1.0f / 128.0f);
        float rs = rsqrtf(q * (1.0f / 128.0f) - mu * mu + 1e-5f);
        for (int c2 = 0; c2 < 128; ++c2) {
            float v = xls[c2 * 256 + t];
            xh[t * XHS + c2] = (__bf16)((v - mu) * rs * lnsS[c2] + lnbS[c2]);
        }
    }
    __syncthreads();

    // ---- Phase 3: (xh @ w_z + b_z) * g ; wave w owns row-tiles {2w, 2w+1} ----
#pragma unroll
    for (int rti = 0; rti < 2; ++rti) {
        const int rt = w * 2 + rti;             // i - i0 == rt for this task
        const __bf16* xrow = xh + (rt * 16 + lM) * XHS;
        v16bf Af[4];
#pragma unroll
        for (int ks = 0; ks < 4; ++ks) Af[ks] = load_afrag(xrow, ks * 32, hA);
        const int i = i0 + rt;
#pragma unroll 1
        for (int nt = 0; nt < 8; ++nt) {
            const int c2 = nt * 16 + lM;
            const __bf16* WCol = wzT + (size_t)c2 * 128;
            v8f acc = {};
#pragma unroll
            for (int ks = 0; ks < 4; ++ks)
                acc = wmma_bf16(Af[ks], load_bfrag(WCol, ks * 32, hB), acc);
            const float bz = b_z[c2];
#pragma unroll
            for (int r = 0; r < 8; ++r) {
                const int j = j0 + r + hi * 8;
                const size_t idx = ((size_t)i * N + j) * C + c2;
                out[idx] = (acc[r] + bz) * gws[idx];
            }
        }
    }
}

// ---------------------------------------------------------------------------
extern "C" void kernel_launch(void* const* d_in, const int* in_sizes, int n_in,
                              void* d_out, int out_size, void* d_ws, size_t ws_size,
                              hipStream_t stream) {
    const float* z      = (const float*)d_in[0];
    const float* mask   = (const float*)d_in[1];
    const float* w_a_p  = (const float*)d_in[2];
    const float* b_a_p  = (const float*)d_in[3];
    const float* w_a_g  = (const float*)d_in[4];
    const float* b_a_g  = (const float*)d_in[5];
    const float* w_b_p  = (const float*)d_in[6];
    const float* b_b_p  = (const float*)d_in[7];
    const float* w_b_g  = (const float*)d_in[8];
    const float* b_b_g  = (const float*)d_in[9];
    const float* w_g    = (const float*)d_in[10];
    const float* b_g    = (const float*)d_in[11];
    const float* w_z    = (const float*)d_in[12];
    const float* b_z    = (const float*)d_in[13];
    const float* ln_i_s = (const float*)d_in[14];
    const float* ln_i_b = (const float*)d_in[15];
    const float* ln_o_s = (const float*)d_in[16];
    const float* ln_o_b = (const float*)d_in[17];

    char* ws = (char*)d_ws;
    __bf16* wT  = (__bf16*)ws;                                    // 196,608 B
    __bf16* aT  = (__bf16*)(ws + 262144);                         // 64 MB
    __bf16* bT  = (__bf16*)(ws + 262144 + 67108864);              // 64 MB
    float*  gws = (float*)(ws + 262144 + 2ull * 67108864);        // 128 MB

    // 0) weights -> bf16 [out][in]:  a_p, a_g, b_p, b_g, g, z
    k_prep<<<6, 256, 0, stream>>>(w_a_p, w_a_g, w_b_p, w_b_g, w_g, w_z, wT);

    // 1) fused LN + projections + gates
    k_ln_proj<<<16384, 256, 0, stream>>>(z, mask, b_a_p, b_a_g, b_b_p, b_b_g,
                                         b_g, ln_i_s, ln_i_b, wT, aT, bT, gws);

    // 2) fused triangle einsum + LN + final projection + output gate
    const size_t smem2 = 128 * 256 * 4 + 256 * 136 * 2 + 256 * 4;  // 201,728 B
    hipFuncSetAttribute((const void*)k_tri,
                        hipFuncAttributeMaxDynamicSharedMemorySize, (int)smem2);
    dim3 g2(32, 32);
    k_tri<<<g2, 256, smem2, stream>>>(aT, bT, gws, wT + 5ull * 16384,
                                      b_z, ln_o_s, ln_o_b, (float*)d_out);
}
